// SpatialContextAwareLocalAttention_46016279609355
// MI455X (gfx1250) — compile-verified, matrix operand order checked by reading, not run on hardware
//
#include <hip/hip_runtime.h>
#include <hip/hip_bf16.h>

// ---------------- problem constants ----------------
#define DIM 256
#define HEADS 8
#define HEAD_DIM 32
#define NT 49                    // tokens per 7x7 window
#define HGT 112
#define WID 112
#define BATCH 8
#define HW (HGT * WID)           // 12544
#define CHW (DIM * HW)           // 3211264
#define TOK (BATCH * HW)         // 100352 tokens
#define QKV_COLS (3 * DIM)       // 768
#define ATT_SCALE 0.17677669529663687f   // 32^-0.5

typedef __attribute__((ext_vector_type(16))) __bf16 v16bf;
typedef __attribute__((ext_vector_type(8)))  float  v8f;

union FragAB { unsigned int u[8]; v16bf b; };
union FragC  { v8f v; float f[8]; };

// native bf16 convert (RNE) -> v_cvt_*bf16_f32 instead of ALU emulation
__device__ __forceinline__ unsigned short f2bf(float f) {
  __bf16 h = (__bf16)f;
  return __builtin_bit_cast(unsigned short, h);
}

// K index pattern for 16-bit A/B fragments (ISA 7.12.2):
// lanes 0-15 (hi=0): VGPR0..3 -> K 0..7 pairs, VGPR4..7 -> K 16..23
// lanes 16-31 (hi=1): VGPR0..3 -> K 8..15,     VGPR4..7 -> K 24..31
__device__ __forceinline__ int kpat(int i, int hi) {
  return ((i & 4) ? 16 : 0) + hi * 8 + (i & 3) * 2;
}

// =====================================================================
// Kernel A: qkv[t][768] (bf16) = X_token(t) @ Wqkv + b_qkv
// 64x64 output tile per block, 8 waves x 2 wmma tiles, K=256.
// =====================================================================
__global__ __launch_bounds__(256) void qkv_kernel(
    const float* __restrict__ x, const float* __restrict__ w_qkv,
    const float* __restrict__ b_qkv, unsigned short* __restrict__ qkv) {
  __shared__ unsigned short XS[64][40];   // [token][k] bf16 bits
  __shared__ unsigned short WT[64][40];   // [col j][k] bf16 bits (B^T)
  const int tid = threadIdx.x;
  const int lane = tid & 31;
  const int wv = tid >> 5;
  const int hi = lane >> 4;
  const int nl = lane & 15;
  const int rt = blockIdx.x / 12;
  const int ct = blockIdx.x % 12;
  const int t0 = rt * 64;
  const int j0 = ct * 64;

  const int tl = tid & 63;                       // token within tile (loads)
  const int tglob = t0 + tl;
  const int xb = (tglob / HW) * CHW + (tglob % HW);  // &x[b][0][h][w]
  const int cg = tid >> 6;                       // 0..3
  const int jl = tid & 63;

  const int rtl = wv >> 1;                       // row tile 0..3
  const int cp = (wv & 1) * 2;                   // col tile pair base

  FragC acc0, acc1;
#pragma unroll
  for (int r = 0; r < 8; ++r) { acc0.f[r] = 0.0f; acc1.f[r] = 0.0f; }

  for (int k0 = 0; k0 < DIM; k0 += 32) {
#pragma unroll
    for (int p = 0; p < 8; ++p) {
      int co = cg + p * 4;                       // 0..31
      int c = k0 + co;
      XS[tl][co] = f2bf(x[xb + c * HW]);                 // coalesced along tokens
      WT[jl][co] = f2bf(w_qkv[c * QKV_COLS + j0 + jl]);  // coalesced along j
    }
    __syncthreads();
    FragAB a, b0, b1;
#pragma unroll
    for (int i = 0; i < 8; ++i) {
      int k = kpat(i, hi);
      a.u[i]  = *(const unsigned int*)&XS[rtl * 16 + nl][k];
      b0.u[i] = *(const unsigned int*)&WT[(cp + 0) * 16 + nl][k];
      b1.u[i] = *(const unsigned int*)&WT[(cp + 1) * 16 + nl][k];
    }
    acc0.v = __builtin_amdgcn_wmma_f32_16x16x32_bf16(false, a.b, false, b0.b,
                                                     (short)0, acc0.v, false, false);
    acc1.v = __builtin_amdgcn_wmma_f32_16x16x32_bf16(false, a.b, false, b1.b,
                                                     (short)0, acc1.v, false, false);
    __syncthreads();
  }
#pragma unroll
  for (int tile = 0; tile < 2; ++tile) {
    FragC& acc = tile ? acc1 : acc0;
    int j = j0 + (cp + tile) * 16 + nl;
    float bias = b_qkv[j];
#pragma unroll
    for (int r = 0; r < 8; ++r) {               // C/D: lane=col, VGPR r -> row r+hi*8
      int t = t0 + rtl * 16 + r + hi * 8;
      qkv[(size_t)t * QKV_COLS + j] = f2bf(acc.f[r] + bias);
    }
  }
}

// =====================================================================
// Kernel B: one wave per (window, head). 49 tokens padded to 64.
// S = (Q K^T)*scale + rpb  -> softmax -> O = P V, stored bf16 to obuf.
// =====================================================================
__global__ __launch_bounds__(32) void attn_kernel(
    const unsigned short* __restrict__ qkv, const float* __restrict__ rpb,
    unsigned short* __restrict__ obuf) {
  __shared__ unsigned short VT[32][72];   // V transposed: [d][token]
  __shared__ unsigned short PS[64][72];   // P: [row m][col n] bf16
  __shared__ float biasH[169];            // per-head rel-pos bias table
  const int lane = threadIdx.x;
  const int hi = lane >> 4;
  const int nl = lane & 15;
  const int win = blockIdx.x >> 3;
  const int head = blockIdx.x & 7;
  const int b = win >> 8;
  const int rem = win & 255;
  const int hn = rem >> 4;
  const int wn = rem & 15;
  const int wbase = b * HW + hn * 7 * WID + wn * 7;  // token id of window (0,0)

  for (int idx = lane; idx < 169; idx += 32)
    biasH[idx] = rpb[idx * HEADS + head];

  { // stage V^T (zero-padded to 64 tokens)
    const int d = lane;                          // 0..31
    int mt = 0;
    for (int i = 0; i < 7; ++i)
      for (int j = 0; j < 7; ++j) {
        int t = wbase + i * WID + j;
        VT[d][mt] = qkv[(size_t)t * QKV_COLS + 512 + head * 32 + d];
        ++mt;
      }
    for (; mt < 64; ++mt) VT[d][mt] = 0;
  }

  // Q as A-fragments, K as B-fragments (head_dim = 32 = exactly one WMMA K)
  FragAB qf[4], kf[4];
#pragma unroll
  for (int mt = 0; mt < 4; ++mt) {
    int row = mt * 16 + nl;
    if (row < NT) {
      int i = row / 7, j = row - i * 7;
      int t = wbase + i * WID + j;
      int baseq = t * QKV_COLS + head * 32;
      int basek = t * QKV_COLS + 256 + head * 32;
#pragma unroll
      for (int e = 0; e < 8; ++e) {
        int k = kpat(e, hi);
        qf[mt].u[e] = *(const unsigned int*)(qkv + baseq + k);
        kf[mt].u[e] = *(const unsigned int*)(qkv + basek + k);
      }
    } else {
#pragma unroll
      for (int e = 0; e < 8; ++e) { qf[mt].u[e] = 0u; kf[mt].u[e] = 0u; }
    }
  }
  __syncthreads();

  float ls[4][8];                                // row sums of exp
#pragma unroll
  for (int mt = 0; mt < 4; ++mt) {
    FragC s[4];
#pragma unroll
    for (int nt = 0; nt < 4; ++nt) {
#pragma unroll
      for (int r = 0; r < 8; ++r) s[nt].f[r] = 0.0f;
      s[nt].v = __builtin_amdgcn_wmma_f32_16x16x32_bf16(false, qf[mt].b, false, kf[nt].b,
                                                        (short)0, s[nt].v, false, false);
    }
    // scale + relative-position bias + column mask
#pragma unroll
    for (int nt = 0; nt < 4; ++nt) {
      int n = nt * 16 + nl;
      int ni = n / 7, nj = n - ni * 7;
      bool nv = (n < NT);
#pragma unroll
      for (int r = 0; r < 8; ++r) {
        int m = mt * 16 + r + hi * 8;
        float v = s[nt].f[r] * ATT_SCALE;
        if (nv && m < NT) {
          int mi = m / 7, mj = m - mi * 7;
          v += biasH[(mi - ni + 6) * 13 + (mj - nj + 6)];
        } else {
          v = -1e30f;
        }
        s[nt].f[r] = v;
      }
    }
    // softmax: row max / exp / row sum (16-lane reduction = column dimension)
    float mx[8], sm[8];
#pragma unroll
    for (int r = 0; r < 8; ++r) {
      float m0 = fmaxf(fmaxf(s[0].f[r], s[1].f[r]), fmaxf(s[2].f[r], s[3].f[r]));
#pragma unroll
      for (int off = 1; off <= 8; off <<= 1)
        m0 = fmaxf(m0, __shfl_xor(m0, off, 32));
      mx[r] = m0;
      sm[r] = 0.0f;
    }
#pragma unroll
    for (int nt = 0; nt < 4; ++nt) {
#pragma unroll
      for (int r = 0; r < 8; ++r) {
        float p = __expf(s[nt].f[r] - mx[r]);
        sm[r] += p;
        PS[mt * 16 + r + hi * 8][nt * 16 + nl] = f2bf(p);
      }
    }
#pragma unroll
    for (int r = 0; r < 8; ++r) {
      float t = sm[r];
#pragma unroll
      for (int off = 1; off <= 8; off <<= 1)
        t += __shfl_xor(t, off, 32);
      ls[mt][r] = t;
    }
  }
  __syncthreads();

  // O = P @ V (K = 64 -> 2 wmma K-steps), normalize, store bf16
#pragma unroll
  for (int mt = 0; mt < 4; ++mt) {
#pragma unroll
    for (int dt = 0; dt < 2; ++dt) {
      FragC o;
#pragma unroll
      for (int r = 0; r < 8; ++r) o.f[r] = 0.0f;
#pragma unroll
      for (int kk = 0; kk < 64; kk += 32) {
        FragAB pa, vb;
#pragma unroll
        for (int e = 0; e < 8; ++e) {
          int k = kk + kpat(e, hi);
          pa.u[e] = *(const unsigned int*)&PS[mt * 16 + nl][k];
          vb.u[e] = *(const unsigned int*)&VT[dt * 16 + nl][k];
        }
        o.v = __builtin_amdgcn_wmma_f32_16x16x32_bf16(false, pa.b, false, vb.b,
                                                      (short)0, o.v, false, false);
      }
#pragma unroll
      for (int r = 0; r < 8; ++r) {
        int m = mt * 16 + r + hi * 8;
        if (m < NT) {
          int i = m / 7, j = m - i * 7;
          int t = wbase + i * WID + j;
          obuf[(size_t)t * DIM + head * 32 + dt * 16 + nl] = f2bf(o.f[r] / ls[mt][r]);
        }
      }
    }
  }
}

// =====================================================================
// Kernel C: out(NCHW fp32) = O(bf16) @ Wproj + b_proj.
// A-tile staged with CDNA5 async global->LDS DMA (ASYNCcnt), LDS-transposed
// epilogue for coalesced NCHW scatter.
// =====================================================================
__global__ __launch_bounds__(256) void proj_kernel(
    const unsigned short* __restrict__ obuf, const float* __restrict__ w_proj,
    const float* __restrict__ b_proj, float* __restrict__ out) {
  __shared__ unsigned short XS[64][40];
  __shared__ unsigned short WT[64][40];
  __shared__ float OT[64][68];            // [token][j] for transposed store
  const int tid = threadIdx.x;
  const int lane = tid & 31;
  const int wv = tid >> 5;
  const int hi = lane >> 4;
  const int nl = lane & 15;
  const int rt = blockIdx.x >> 2;
  const int ct = blockIdx.x & 3;
  const int t0 = rt * 64;
  const int j0 = ct * 64;
  const int rtl = wv >> 1;
  const int cp = (wv & 1) * 2;

  const int arow = tid >> 2;              // 0..63
  const int acg = tid & 3;                // 0..3 -> 8 bf16 each
  const int cg = tid >> 6;
  const int jl = tid & 63;

  // per-lane LDS byte offset for the async DMA destination (generic LDS
  // address truncates to the raw LDS offset per the ISA aperture mapping)
  const unsigned lds_dst = (unsigned)(size_t)&XS[arow][acg * 8];
  const unsigned short* gsrc_base = obuf + (size_t)(t0 + arow) * DIM + acg * 8;

  FragC acc0, acc1;
#pragma unroll
  for (int r = 0; r < 8; ++r) { acc0.f[r] = 0.0f; acc1.f[r] = 0.0f; }

  for (int k0 = 0; k0 < DIM; k0 += 32) {
    // A tile: 16B async DMA straight into LDS (no VGPR round-trip)
    {
      const unsigned short* gsrc = gsrc_base + k0;
      asm volatile("global_load_async_to_lds_b128 %0, %1, off"
                   :: "v"(lds_dst), "v"(gsrc) : "memory");
    }
#pragma unroll
    for (int p = 0; p < 8; ++p) {
      int co = cg + p * 4;
      WT[jl][co] = f2bf(w_proj[(k0 + co) * DIM + j0 + jl]);
    }
    asm volatile("s_wait_asynccnt 0x0" ::: "memory");
    __syncthreads();
    FragAB a, b0, b1;
#pragma unroll
    for (int i = 0; i < 8; ++i) {
      int k = kpat(i, hi);
      a.u[i]  = *(const unsigned int*)&XS[rtl * 16 + nl][k];
      b0.u[i] = *(const unsigned int*)&WT[(cp + 0) * 16 + nl][k];
      b1.u[i] = *(const unsigned int*)&WT[(cp + 1) * 16 + nl][k];
    }
    acc0.v = __builtin_amdgcn_wmma_f32_16x16x32_bf16(false, a.b, false, b0.b,
                                                     (short)0, acc0.v, false, false);
    acc1.v = __builtin_amdgcn_wmma_f32_16x16x32_bf16(false, a.b, false, b1.b,
                                                     (short)0, acc1.v, false, false);
    __syncthreads();
  }
#pragma unroll
  for (int tile = 0; tile < 2; ++tile) {
    FragC& acc = tile ? acc1 : acc0;
    int jloc = (cp + tile) * 16 + nl;
    float bias = b_proj[j0 + jloc];
#pragma unroll
    for (int r = 0; r < 8; ++r) {
      int m = rtl * 16 + r + hi * 8;
      OT[m][jloc] = acc.f[r] + bias;
    }
  }
  __syncthreads();
  // scatter: consecutive tids -> consecutive tokens (contiguous along W)
  const int tl = tid & 63;
  const int t = t0 + tl;
  const int gb = (t / HW) * CHW + (t % HW);
#pragma unroll
  for (int p = 0; p < 16; ++p) {
    int jloc = (tid >> 6) + p * 4;
    out[gb + (j0 + jloc) * HW] = OT[tl][jloc];
  }
}

// =====================================================================
extern "C" void kernel_launch(void* const* d_in, const int* in_sizes, int n_in,
                              void* d_out, int out_size, void* d_ws, size_t ws_size,
                              hipStream_t stream) {
  const float* x      = (const float*)d_in[0];
  const float* w_qkv  = (const float*)d_in[1];
  const float* b_qkv  = (const float*)d_in[2];
  const float* w_proj = (const float*)d_in[3];
  const float* b_proj = (const float*)d_in[4];
  const float* rpb    = (const float*)d_in[5];
  // d_in[6..11] = MCK branch params: computed-but-unused in the reference output.
  float* out = (float*)d_out;

  // workspace: qkv bf16 [TOK][768] (147 MB, lives in 192MB L2 between kernels)
  //            obuf bf16 [TOK][256] (49 MB)
  unsigned short* qkv  = (unsigned short*)d_ws;
  unsigned short* obuf = qkv + (size_t)TOK * QKV_COLS;

  qkv_kernel <<< (TOK / 64) * (QKV_COLS / 64), 256, 0, stream >>>(x, w_qkv, b_qkv, qkv);
  attn_kernel<<< 2048 * HEADS, 32, 0, stream >>>(qkv, rpb, obuf);
  proj_kernel<<< (TOK / 64) * (DIM / 64), 256, 0, stream >>>(obuf, w_proj, b_proj, out);
}
